// MLA_47519518163285
// MI455X (gfx1250) — compile-verified, matrix operand order checked by reading, not run on hardware
//
#include <hip/hip_runtime.h>
#include <stdint.h>

#define SEQ      2048
#define BATCH    2
#define ROWS     (BATCH*SEQ)        // 4096
#define DM       2048               // d_model
#define NH       16
#define HD       128                // head dim (nope)
#define RD       64                 // rope dim
#define QLORA    768
#define KVLORA   512

typedef __attribute__((ext_vector_type(16))) __bf16 v16bf;
typedef __attribute__((ext_vector_type(8)))  float  v8f;

__device__ __forceinline__ unsigned short f2bf(float f) {
    union { float f; unsigned int u; } x; x.f = f;
    unsigned int r = x.u + 0x7FFFu + ((x.u >> 16) & 1u);   // RNE
    return (unsigned short)(r >> 16);
}
__device__ __forceinline__ float bf2f(unsigned short b) {
    union { unsigned int u; float f; } x; x.u = ((unsigned int)b) << 16;
    return x.f;
}

union FragU { v16bf v; uint4 q[2]; };

// A-frag (16x32, row-major [M][K]) and B-frag (32x16, from Bt[N][K] row-major)
// share the same per-lane layout on CDNA5:
//   lane holds row/col (lane&15); element j: K = k0 + (j/8)*16 + (lane>=16?8:0) + j%8
__device__ __forceinline__ v16bf load_frag(const unsigned short* p0, int rowStride,
                                           int k0, int lane) {
    const unsigned short* p = p0 + (size_t)(lane & 15) * rowStride
                                 + k0 + ((lane >> 4) & 1) * 8;
    FragU f;
    f.q[0] = *reinterpret_cast<const uint4*>(p);
    f.q[1] = *reinterpret_cast<const uint4*>(p + 16);
    return f.v;
}

// same pattern but from an LDS tile stored [rows][32] (k contiguous)
__device__ __forceinline__ v16bf lds_frag(const unsigned short* base, int row0, int lane) {
    const unsigned short* p = base + (row0 + (lane & 15)) * 32 + ((lane >> 4) & 1) * 8;
    FragU f;
    f.q[0] = *reinterpret_cast<const uint4*>(p);
    f.q[1] = *reinterpret_cast<const uint4*>(p + 16);
    return f.v;
}

__device__ __forceinline__ v8f wmma_bf16(v16bf a, v16bf b, v8f c) {
    return __builtin_amdgcn_wmma_f32_16x16x32_bf16(false, a, false, b,
                                                   (short)0, c, false, false);
}

// CDNA5 async global->LDS copy of one 16B chunk (ASYNCcnt-tracked, no VGPR data).
// ldsDst: generic pointer to __shared__; low 32 bits of the flat address are the
// LDS byte offset (ISA 10.2 aperture mapping).
__device__ __forceinline__ void async_b128(unsigned short* ldsDst,
                                           const unsigned short* gsrc) {
    unsigned lo = (unsigned)(uintptr_t)ldsDst;
    unsigned long long ga = (unsigned long long)(uintptr_t)gsrc;
    asm volatile("global_load_async_to_lds_b128 %0, %1, off"
                 :: "v"(lo), "v"(ga) : "memory");
}
__device__ __forceinline__ void wait_async0() {
    asm volatile("s_wait_asynccnt 0x0" ::: "memory");
}

// ---------------------------------------------------------------- elementwise
__global__ void k_f32_to_bf16(const float* __restrict__ in,
                              unsigned short* __restrict__ out, long n) {
    long i = (long)blockIdx.x * blockDim.x + threadIdx.x;
    if (i < n) out[i] = f2bf(in[i]);
}

// in f32 [R][C] -> out bf16 [C][R]
__global__ void k_transpose_bf16(const float* __restrict__ in,
                                 unsigned short* __restrict__ out, int R, int C) {
    long i = (long)blockIdx.x * blockDim.x + threadIdx.x;
    if (i >= (long)R * C) return;
    int r = (int)(i / C), c = (int)(i % C);
    out[(size_t)c * R + r] = f2bf(in[i]);
}

// RMSNorm: in f32 [rows][D] * rsqrt(mean(x^2)+eps) * g -> bf16
__global__ void k_rmsnorm(const float* __restrict__ in, const float* __restrict__ g,
                          unsigned short* __restrict__ out, int D) {
    int row = blockIdx.x;
    const float* x = in + (size_t)row * D;
    float ss = 0.f;
    for (int i = threadIdx.x; i < D; i += blockDim.x) { float v = x[i]; ss += v * v; }
    __shared__ float red[256];
    red[threadIdx.x] = ss;
    __syncthreads();
    for (int off = 128; off > 0; off >>= 1) {
        if ((int)threadIdx.x < off) red[threadIdx.x] += red[threadIdx.x + off];
        __syncthreads();
    }
    float sc = rsqrtf(red[0] / (float)D + 1e-6f);
    for (int i = threadIdx.x; i < D; i += blockDim.x)
        out[(size_t)row * D + i] = f2bf(x[i] * sc * g[i]);
}

// RoPE in-place on bf16 [ROWS][nh*64]; position = row % SEQ
__global__ void k_rope(unsigned short* __restrict__ t, int nh) {
    long i = (long)blockIdx.x * blockDim.x + threadIdx.x;
    long total = (long)ROWS * nh * 32;
    if (i >= total) return;
    int  j   = (int)(i & 31);
    int  hh  = (int)((i >> 5) % nh);
    long row = i / (32L * nh);
    int  pos = (int)(row % SEQ);
    float inv = __powf(10000.f, -((float)j) / 32.f);
    float ang = (float)pos * inv;
    float c = __cosf(ang), s = __sinf(ang);
    size_t base = (size_t)row * (nh * 64) + hh * 64;
    float t1 = bf2f(t[base + j]), t2 = bf2f(t[base + j + 32]);
    t[base + j]      = f2bf(t1 * c - t2 * s);
    t[base + j + 32] = f2bf(t2 * c + t1 * s);
}

// ------------------------------------------------- direct GEMM (small N only)
// C[M][N] = A[M][K](bf16) * Bt[N][K](bf16); mode 0: f32 out, 1: bf16 out.
__global__ void __launch_bounds__(256)
k_gemm_small(const unsigned short* __restrict__ A, const unsigned short* __restrict__ Bt,
             float* __restrict__ Cf, unsigned short* __restrict__ Cb,
             int M, int N, int K, int mode) {
    int lane = threadIdx.x & 31;
    int wave = threadIdx.x >> 5;
    int nG = N >> 6;
    long wid = (long)blockIdx.x * 8 + wave;
    long total = (long)(M >> 4) * nG;
    if (wid >= total) return;
    int m0 = (int)(wid / nG) << 4;
    int n0 = (int)(wid % nG) << 6;

    v8f z = {0.f,0.f,0.f,0.f,0.f,0.f,0.f,0.f};
    v8f acc[4] = {z, z, z, z};
    const unsigned short* Arow = A + (size_t)m0 * K;

    for (int k0 = 0; k0 < K; k0 += 32) {
        v16bf a = load_frag(Arow, K, k0, lane);
#pragma unroll
        for (int t = 0; t < 4; ++t) {
            v16bf b = load_frag(Bt + (size_t)(n0 + t * 16) * K, K, k0, lane);
            acc[t] = wmma_bf16(a, b, acc[t]);
        }
    }
    int half = (lane >> 4) & 1, nl = lane & 15;
#pragma unroll
    for (int t = 0; t < 4; ++t)
#pragma unroll
        for (int r = 0; r < 8; ++r) {
            int m = m0 + r + 8 * half;
            int n = n0 + t * 16 + nl;
            if (mode == 0) Cf[(size_t)m * N + n] = acc[t][r];
            else           Cb[(size_t)m * N + n] = f2bf(acc[t][r]);
        }
}

// ----------------------------------- LDS-tiled GEMM (async double-buffered)
// Block = 256 thr (8 waves), block tile 128x128, k-step 32.
// A/B 128x32 slices staged to LDS with global_load_async_to_lds_b128;
// each wave computes a 32x64 register tile (8 WMMAs per k-step).
// mode 0: fp32 out | 1: bf16 out | 2: bf16 transposed out [N][M]
__global__ void __launch_bounds__(256)
k_gemm_lds(const unsigned short* __restrict__ A, const unsigned short* __restrict__ Bt,
           float* __restrict__ Cf, unsigned short* __restrict__ Cb,
           int M, int N, int K, int mode) {
    __shared__ __attribute__((aligned(16))) unsigned short bufA[2][128 * 32];
    __shared__ __attribute__((aligned(16))) unsigned short bufB[2][128 * 32];

    int tid  = threadIdx.x;
    int lane = tid & 31;
    int wave = tid >> 5;
    int nBlk = N >> 7;
    int bm = (int)blockIdx.x / nBlk;
    int bn = (int)blockIdx.x % nBlk;
    int m0 = bm << 7, n0 = bn << 7;
    int m0w = (wave & 3) * 32;    // wave tile: 32 (M) x 64 (N)
    int n0w = (wave >> 2) * 64;

    const unsigned short* Ab = A  + (size_t)m0 * K;
    const unsigned short* Bb = Bt + (size_t)n0 * K;

    // stage one 128x32 k-slice of A and B into buffer `buf`
    auto stage = [&](int buf, int kk) {
#pragma unroll
        for (int i = 0; i < 2; ++i) {
            int chunk = tid + i * 256;          // 0..511
            int row = chunk >> 2, q = chunk & 3; // 4 x b128 per 32-elem row
            async_b128(&bufA[buf][row * 32 + q * 8], Ab + (size_t)row * K + kk + q * 8);
            async_b128(&bufB[buf][row * 32 + q * 8], Bb + (size_t)row * K + kk + q * 8);
        }
    };

    v8f z = {0.f,0.f,0.f,0.f,0.f,0.f,0.f,0.f};
    v8f acc[2][4] = {{z, z, z, z}, {z, z, z, z}};

    int nsteps = K >> 5;
    stage(0, 0);
    wait_async0();
    __syncthreads();

    for (int s = 0; s < nsteps; ++s) {
        int cur = s & 1;
        if (s + 1 < nsteps) stage(cur ^ 1, (s + 1) << 5);   // overlap with compute
        v16bf af0 = lds_frag(bufA[cur], m0w,      lane);
        v16bf af1 = lds_frag(bufA[cur], m0w + 16, lane);
#pragma unroll
        for (int t = 0; t < 4; ++t) {
            v16bf bf = lds_frag(bufB[cur], n0w + t * 16, lane);
            acc[0][t] = wmma_bf16(af0, bf, acc[0][t]);
            acc[1][t] = wmma_bf16(af1, bf, acc[1][t]);
        }
        wait_async0();          // next slice resident in LDS
        __syncthreads();        // all waves done reading `cur` before overwrite
    }

    int half = (lane >> 4) & 1, nl = lane & 15;
#pragma unroll
    for (int ms = 0; ms < 2; ++ms)
#pragma unroll
        for (int t = 0; t < 4; ++t)
#pragma unroll
            for (int r = 0; r < 8; ++r) {
                int m = m0 + m0w + ms * 16 + r + 8 * half;
                int n = n0 + n0w + t * 16 + nl;
                float v = acc[ms][t][r];
                if (mode == 0)      Cf[(size_t)m * N + n] = v;
                else if (mode == 1) Cb[(size_t)m * N + n] = f2bf(v);
                else                Cb[(size_t)n * M + m] = f2bf(v);
            }
}

// ---------------------------------------------------------------- attention
// One wave per (b, h, 16-query tile). Flash-style streaming softmax.
__global__ void __launch_bounds__(32)
k_attn(const unsigned short* __restrict__ qn, const unsigned short* __restrict__ qr,
       const unsigned short* __restrict__ kn, const unsigned short* __restrict__ kr,
       const unsigned short* __restrict__ vt, unsigned short* __restrict__ obuf) {
    __shared__ __attribute__((aligned(16))) unsigned short pLds[16 * 32];
    int lane = threadIdx.x & 31;
    int bid = blockIdx.x;
    int qt = bid & 127;          // SEQ/16 tiles
    int h  = (bid >> 7) & 15;
    int b  = bid >> 11;
    int q0 = qt << 4;
    const float scale = 0.07216878365f;   // 1/sqrt(192)
    int half = lane >> 4, nl = lane & 15;

    v16bf qa[6];
    const unsigned short* qnp = qn + (size_t)(b * SEQ + q0) * (NH * HD) + h * HD;
#pragma unroll
    for (int c = 0; c < 4; ++c) qa[c] = load_frag(qnp, NH * HD, c * 32, lane);
    const unsigned short* qrp = qr + (size_t)(b * SEQ + q0) * (NH * RD) + h * RD;
#pragma unroll
    for (int c = 0; c < 2; ++c) qa[4 + c] = load_frag(qrp, NH * RD, c * 32, lane);

    v8f z = {0.f,0.f,0.f,0.f,0.f,0.f,0.f,0.f};
    v8f accO[8] = {z, z, z, z, z, z, z, z};
    float mrow[8], lrow[8];
#pragma unroll
    for (int r = 0; r < 8; ++r) { mrow[r] = -1e30f; lrow[r] = 0.f; }

    for (int key0 = 0; key0 < SEQ; key0 += 32) {
        v8f S[2] = {z, z};
#pragma unroll
        for (int sub = 0; sub < 2; ++sub) {
            int ks0 = key0 + sub * 16;
            const unsigned short* knp = kn + (size_t)(b * SEQ + ks0) * (NH * HD) + h * HD;
#pragma unroll
            for (int c = 0; c < 4; ++c)
                S[sub] = wmma_bf16(qa[c], load_frag(knp, NH * HD, c * 32, lane), S[sub]);
            const unsigned short* krp = kr + (size_t)(b * SEQ + ks0) * RD;
#pragma unroll
            for (int c = 0; c < 2; ++c)
                S[sub] = wmma_bf16(qa[4 + c], load_frag(krp, RD, c * 32, lane), S[sub]);
        }
        float p0[8], p1[8], alpha[8];
#pragma unroll
        for (int r = 0; r < 8; ++r) {
            float s0 = S[0][r] * scale, s1 = S[1][r] * scale;
            float tm = fmaxf(s0, s1);
            tm = fmaxf(tm, __shfl_xor(tm, 1, 32));
            tm = fmaxf(tm, __shfl_xor(tm, 2, 32));
            tm = fmaxf(tm, __shfl_xor(tm, 4, 32));
            tm = fmaxf(tm, __shfl_xor(tm, 8, 32));
            float nm = fmaxf(mrow[r], tm);
            alpha[r] = __expf(mrow[r] - nm);
            mrow[r] = nm;
            p0[r] = __expf(s0 - nm);
            p1[r] = __expf(s1 - nm);
            float rs = p0[r] + p1[r];
            rs += __shfl_xor(rs, 1, 32);
            rs += __shfl_xor(rs, 2, 32);
            rs += __shfl_xor(rs, 4, 32);
            rs += __shfl_xor(rs, 8, 32);
            lrow[r] = lrow[r] * alpha[r] + rs;
        }
#pragma unroll
        for (int t = 0; t < 8; ++t)
#pragma unroll
            for (int r = 0; r < 8; ++r) accO[t][r] *= alpha[r];

        // D-layout -> A-layout for P via LDS (16x32 bf16 tile)
#pragma unroll
        for (int r = 0; r < 8; ++r) {
            int m = r + 8 * half;
            pLds[m * 32 + nl]      = f2bf(p0[r]);
            pLds[m * 32 + nl + 16] = f2bf(p1[r]);
        }
        __syncthreads();
        FragU pf;
        {
            int koff = half * 8;
            pf.q[0] = *reinterpret_cast<const uint4*>(&pLds[nl * 32 + koff]);
            pf.q[1] = *reinterpret_cast<const uint4*>(&pLds[nl * 32 + 16 + koff]);
        }
        __syncthreads();
#pragma unroll
        for (int ft = 0; ft < 8; ++ft) {
            const unsigned short* vp = vt + (size_t)(h * HD + ft * 16) * ROWS + b * SEQ;
            accO[ft] = wmma_bf16(pf.v, load_frag(vp, ROWS, key0, lane), accO[ft]);
        }
    }
#pragma unroll
    for (int r = 0; r < 8; ++r) {
        float inv = 1.f / lrow[r];
        int m = q0 + r + 8 * half;
        size_t base = (size_t)(b * SEQ + m) * (NH * HD) + h * HD;
#pragma unroll
        for (int ft = 0; ft < 8; ++ft)
            obuf[base + ft * 16 + nl] = f2bf(accO[ft][r] * inv);
    }
}

// ---------------------------------------------------------------- host
static inline int cdiv(long a, long b) { return (int)((a + b - 1) / b); }

extern "C" void kernel_launch(void* const* d_in, const int* in_sizes, int n_in,
                              void* d_out, int out_size, void* d_ws, size_t ws_size,
                              hipStream_t stream) {
    const float* x    = (const float*)d_in[0];
    const float* Wcq  = (const float*)d_in[1];
    const float* g_q  = (const float*)d_in[2];
    const float* Wdq  = (const float*)d_in[3];
    const float* Wdqr = (const float*)d_in[4];
    const float* Wckv = (const float*)d_in[5];
    const float* g_kv = (const float*)d_in[6];
    const float* Wdk  = (const float*)d_in[7];
    const float* Wdv  = (const float*)d_in[8];
    const float* Wkr  = (const float*)d_in[9];
    const float* Wo   = (const float*)d_in[10];
    float* out = (float*)d_out;

    char* ws = (char*)d_ws;
    size_t off = 0;
    auto alloc = [&](size_t bytes) -> char* {
        char* p = ws + off;
        off += (bytes + 255) & ~(size_t)255;
        return p;
    };
    typedef unsigned short us;
    float* tq   = (float*)alloc((size_t)ROWS * QLORA  * 4);
    float* tkv  = (float*)alloc((size_t)ROWS * KVLORA * 4);
    us* xb    = (us*)alloc((size_t)ROWS * DM * 2);
    us* WcqT  = (us*)alloc((size_t)QLORA * DM * 2);
    us* WdqT  = (us*)alloc((size_t)(NH * HD) * QLORA * 2);
    us* WdqrT = (us*)alloc((size_t)(NH * RD) * QLORA * 2);
    us* WckvT = (us*)alloc((size_t)KVLORA * DM * 2);
    us* WdkT  = (us*)alloc((size_t)(NH * HD) * KVLORA * 2);
    us* WdvT  = (us*)alloc((size_t)(NH * HD) * KVLORA * 2);
    us* WkrT  = (us*)alloc((size_t)RD * DM * 2);
    us* WoT   = (us*)alloc((size_t)DM * (NH * HD) * 2);
    us* qc    = (us*)alloc((size_t)ROWS * QLORA * 2);
    us* kvc   = (us*)alloc((size_t)ROWS * KVLORA * 2);
    us* qn    = (us*)alloc((size_t)ROWS * NH * HD * 2);
    us* qr    = (us*)alloc((size_t)ROWS * NH * RD * 2);
    us* kn    = (us*)alloc((size_t)ROWS * NH * HD * 2);
    us* vt    = (us*)alloc((size_t)NH * HD * ROWS * 2);
    us* krb   = (us*)alloc((size_t)ROWS * RD * 2);
    us* obuf  = (us*)tq;  // alias: tq+tkv dead before attention writes obuf

    const int B256 = 256;
    {
        long n = (long)ROWS * DM;
        k_f32_to_bf16<<<cdiv(n, B256), B256, 0, stream>>>(x, xb, n);
    }
    k_transpose_bf16<<<cdiv((long)DM * QLORA, B256), B256, 0, stream>>>(Wcq, WcqT, DM, QLORA);
    k_transpose_bf16<<<cdiv((long)QLORA * NH * HD, B256), B256, 0, stream>>>(Wdq, WdqT, QLORA, NH * HD);
    k_transpose_bf16<<<cdiv((long)QLORA * NH * RD, B256), B256, 0, stream>>>(Wdqr, WdqrT, QLORA, NH * RD);
    k_transpose_bf16<<<cdiv((long)DM * KVLORA, B256), B256, 0, stream>>>(Wckv, WckvT, DM, KVLORA);
    k_transpose_bf16<<<cdiv((long)KVLORA * NH * HD, B256), B256, 0, stream>>>(Wdk, WdkT, KVLORA, NH * HD);
    k_transpose_bf16<<<cdiv((long)KVLORA * NH * HD, B256), B256, 0, stream>>>(Wdv, WdvT, KVLORA, NH * HD);
    k_transpose_bf16<<<cdiv((long)DM * RD, B256), B256, 0, stream>>>(Wkr, WkrT, DM, RD);
    k_transpose_bf16<<<cdiv((long)(NH * HD) * DM, B256), B256, 0, stream>>>(Wo, WoT, NH * HD, DM);

    auto gemm = [&](const us* A, const us* Bt, float* Cf, us* Cb,
                    int M, int N, int K, int mode) {
        k_gemm_lds<<<(M >> 7) * (N >> 7), 256, 0, stream>>>(A, Bt, Cf, Cb, M, N, K, mode);
    };
    // latent projections
    gemm(xb, WcqT,  tq,  nullptr, ROWS, QLORA,  DM, 0);
    gemm(xb, WckvT, tkv, nullptr, ROWS, KVLORA, DM, 0);
    {   // x @ Wkr (N=64): direct kernel
        long tiles = (long)(ROWS >> 4) * (RD >> 6);
        k_gemm_small<<<cdiv(tiles, 8), 256, 0, stream>>>(xb, WkrT, nullptr, krb,
                                                         ROWS, RD, DM, 1);
    }
    k_rmsnorm<<<ROWS, 256, 0, stream>>>(tq,  g_q,  qc,  QLORA);
    k_rmsnorm<<<ROWS, 256, 0, stream>>>(tkv, g_kv, kvc, KVLORA);
    // head projections
    gemm(qc,  WdqT,  nullptr, qn, ROWS, NH * HD, QLORA,  1);
    gemm(qc,  WdqrT, nullptr, qr, ROWS, NH * RD, QLORA,  1);
    gemm(kvc, WdkT,  nullptr, kn, ROWS, NH * HD, KVLORA, 1);
    gemm(kvc, WdvT,  nullptr, vt, ROWS, NH * HD, KVLORA, 2);  // transposed store
    // rope
    k_rope<<<cdiv((long)ROWS * NH * 32, B256), B256, 0, stream>>>(qr, NH);
    k_rope<<<cdiv((long)ROWS * 1 * 32, B256), B256, 0, stream>>>(krb, 1);
    // attention
    k_attn<<<BATCH * NH * (SEQ / 16), 32, 0, stream>>>(qn, qr, kn, krb, vt, obuf);
    // output projection
    gemm(obuf, WoT, out, nullptr, ROWS, DM, NH * HD, 0);
}
